// VectorQuantizer_55430847922266
// MI455X (gfx1250) — matrix operand order repaired and checked by reading, not compiled, not run on hardware
//
#include <hip/hip_runtime.h>

#define NUM_EMB 1024
#define EMB_DIM 256
#define HW      4096        // 64*64
#define TILE_M  128
#define PAD_ROW 264         // halves per padded LDS row (256 + 8) -> bank-conflict-free
#define LOSS_OFF 16777216
#define IDX_OFF  16777217

typedef __attribute__((ext_vector_type(16))) _Float16 v16h;
typedef __attribute__((ext_vector_type(8)))  _Float16 v8h;
typedef __attribute__((ext_vector_type(8)))  float    v8f;
typedef __attribute__((ext_vector_type(4)))  unsigned int u32x4;
typedef __attribute__((ext_vector_type(8)))  int      i32x8;
typedef __attribute__((ext_vector_type(4)))  int      i32x4;

// ---------------- prep: f32 codebook -> f16 + code norms + zero accumulator --
__global__ void vq_prep(const float* __restrict__ cb,
                        _Float16* __restrict__ cbh,
                        float* __restrict__ norms,
                        float* __restrict__ sumsq) {
    const int k = blockIdx.x;         // code index 0..1023
    const int t = threadIdx.x;        // dim index 0..255
    float v = cb[k * EMB_DIM + t];
    cbh[k * EMB_DIM + t] = (_Float16)v;
    float p = v * v;
    #pragma unroll
    for (int m = 16; m > 0; m >>= 1) p += __shfl_xor(p, m, 32);
    __shared__ float ws[8];
    if ((t & 31) == 0) ws[t >> 5] = p;
    __syncthreads();
    if (t == 0) {
        float s = 0.f;
        #pragma unroll
        for (int i = 0; i < 8; ++i) s += ws[i];
        norms[k] = s;
        if (k == 0) *sumsq = 0.f;
    }
}

// load 16 contiguous f16 from LDS as two 16B chunks (16B-aligned)
__device__ inline v16h ldv16(const _Float16* p) {
    v8h lo = *(const v8h*)p;
    v8h hi = *(const v8h*)(p + 8);
    return __builtin_shufflevector(lo, hi, 0,1,2,3,4,5,6,7,8,9,10,11,12,13,14,15);
}

// TDM: DMA one 64KB codebook group (128 codes x 512B) global->LDS, inserting
// 16B of padding every 512B (pad_interval=6 -> 128 DWORDs, pad_amount=3 -> 4 DWORDs)
// so LDS rows land at the bank-conflict-free PAD_ROW=264-half stride.
__device__ inline void tdm_load_group(const _Float16* src, _Float16* dstLds) {
    const unsigned lds_off = (unsigned)(size_t)dstLds;   // low 32 bits of generic ptr = LDS offset
    const unsigned long long ga = (unsigned long long)(size_t)src;
    u32x4 g0;
    g0.x = 1u;                                            // count=1 user descriptor
    g0.y = lds_off;                                       // lds_addr
    g0.z = (unsigned)(ga & 0xFFFFFFFFu);                  // global_addr[31:0]
    g0.w = (unsigned)((ga >> 32) & 0x01FFFFFFu) | (2u << 30);  // global_addr[56:32], type=2
    i32x8 g1;
    g1[0] = (int)((2u << 16) | (1u << 20) | (6u << 22) | (3u << 25)); // data_size=4B, pad_en, interval, amount
    g1[1] = (int)(16384u << 16);   // tensor_dim0 = 16384 DWORDs (1-D view of the 64KB group)
    g1[2] = (int)(1u << 16);       // tensor_dim0 hi = 0; tensor_dim1 = 1
    g1[3] = (int)(16384u << 16);   // tensor_dim1 hi = 0; tile_dim0 = 16384 DWORDs
    g1[4] = 0;                     // tile_dim1 = 0 (unused), tile_dim2 = 0
    g1[5] = 16384;                 // tensor_dim0_stride
    g1[6] = 0;
    g1[7] = 0;
    i32x4 gz = {0, 0, 0, 0};
#if __clang_major__ >= 23
    i32x8 gz8 = {};
    __builtin_amdgcn_tensor_load_to_lds(g0, g1, gz, gz, gz8, 0);
#else
    __builtin_amdgcn_tensor_load_to_lds(g0, g1, gz, gz, 0);
#endif
}

// ---------------- main: WMMA distance argmin + gather/writeback --------------
__launch_bounds__(256)
__global__ void vq_main(const float* __restrict__ in,      // [16,256,64,64]
                        const float* __restrict__ cb,      // [1024,256] f32
                        const _Float16* __restrict__ cbh,  // [1024,256] f16
                        const float* __restrict__ norms,   // [1024]
                        float* __restrict__ out,
                        float* __restrict__ sumsq) {
    __shared__ _Float16 cbs[2][TILE_M * PAD_ROW];  // double-buffered codebook groups
    __shared__ float    nlds[NUM_EMB];             // code norms
    __shared__ int      idx_lds[TILE_M];
    __shared__ float    wsum[8];

    const int tid  = threadIdx.x;
    const int lane = tid & 31;
    const int wave = tid >> 5;
    const int nlane = lane & 15;           // N within 16x16 tile
    const int hi8   = (lane >> 4) * 8;     // unused (A fragment also uses kh16)
    const int kh16  = (lane >> 4) * 16;    // B-fragment K sub-offset

    const long nb = (long)blockIdx.x * TILE_M;          // base flat row
    const int  bb = (int)(nb >> 12);                    // batch
    const int  hw = (int)(nb & 4095);
    const size_t in_base = (size_t)bb * EMB_DIM * HW + (size_t)hw;

    // kick off DMA of codebook group 0 ASAP (overlaps A-fragment loads)
    if (wave == 0) tdm_load_group(cbh, &cbs[0][0]);

    // stage code norms
    #pragma unroll
    for (int i = 0; i < 4; ++i) nlds[i * 256 + tid] = norms[i * 256 + tid];

    // ---- preload this wave's A fragments (16 rows x 256 K) from global ----
    const int r = wave * 16 + nlane;                    // row within tile
    const float* xrow = in + in_base + r;
    v16h afrag[8];
    #pragma unroll
    for (int kk = 0; kk < 8; ++kk) {
        v16h a;
        #pragma unroll
        for (int e = 0; e < 16; ++e) {
            const int c = kk * 32 + kh16 + e;  // K index
            a[e] = (_Float16)xrow[(size_t)c * HW];
        }
        afrag[kk] = a;
    }

    float bestv[8];
    int   besti[8];
    #pragma unroll
    for (int v = 0; v < 8; ++v) { bestv[v] = 3.4e38f; besti[v] = 0; }

    // ---- loop over 8 groups of 128 codes, TDM double-buffered ----
    for (int g = 0; g < 8; ++g) {
        __syncthreads();   // all waves done with buffer (g+1)&1 from iteration g-1
        if (wave == 0) {
            if (g < 7) {
                tdm_load_group(cbh + (size_t)(g + 1) * 128 * EMB_DIM, &cbs[(g + 1) & 1][0]);
                __builtin_amdgcn_s_wait_tensorcnt(1);  // in-order: group g has landed
            } else {
                __builtin_amdgcn_s_wait_tensorcnt(0);
            }
        }
        __syncthreads();   // buffer g&1 visible to all waves
        const _Float16* bufp = &cbs[g & 1][0] + nlane * PAD_ROW + kh16;

        // explicit distance-2 software pipeline on B fragments: the load that
        // feeds WMMA t is issued at t-2 => s_wait_dscnt <= 2, loads overlap WMMA
        v16h b0 = ldv16(bufp + 0);                       // (cbl=0, kk=0)
        v16h b1 = ldv16(bufp + 32);                      // (cbl=0, kk=1)
        #pragma unroll
        for (int cbl = 0; cbl < 8; ++cbl) {
            v8f acc = {};
            #pragma unroll
            for (int kk = 0; kk < 8; ++kk) {
                const v16h bn = b1;
                const int ft = cbl * 8 + kk + 2;         // flattened prefetch slot
                if (ft < 64) {
                    b1 = ldv16(bufp + (ft >> 3) * (16 * PAD_ROW) + (ft & 7) * 32);
                }
                acc = __builtin_amdgcn_wmma_f32_16x16x32_f16(
                        false, afrag[kk], false, b0, (short)0, acc, false, false);
                b0 = bn;
            }
            const int codeidx = (g * 8 + cbl) * 16 + nlane;
            const float cn = nlds[codeidx];
            #pragma unroll
            for (int v = 0; v < 8; ++v) {
                const float d = cn - 2.0f * acc[v];      // ||e||^2 - 2 x.e
                if (d < bestv[v]) { bestv[v] = d; besti[v] = codeidx; }
            }
        }
    }

    // ---- cross-lane argmin over the 16 code-lanes (ties -> smallest index) ----
    #pragma unroll
    for (int v = 0; v < 8; ++v) {
        float bv = bestv[v]; int bi = besti[v];
        #pragma unroll
        for (int m = 1; m < 16; m <<= 1) {
            const float ov = __shfl_xor(bv, m, 32);
            const int   oi = __shfl_xor(bi, m, 32);
            if (ov < bv || (ov == bv && oi < bi)) { bv = ov; bi = oi; }
        }
        if (nlane == 0) idx_lds[wave * 16 + v + (lane >> 4) * 8] = bi;
    }
    __syncthreads();

    // ---- indices output (as float, matching harness output dtype) ----
    if (tid < TILE_M) out[IDX_OFF + nb + tid] = (float)idx_lds[tid];

    // ---- gather code vectors, write quantized (== straight-through fwd), loss ----
    float sq = 0.f;
    #pragma unroll 4
    for (int i = 0; i < 128; ++i) {
        const int linear = i * 256 + tid;   // 0..32767
        const int rr = linear & 127;
        const int cc = linear >> 7;
        const int k  = idx_lds[rr];
        const float q = cb[(size_t)k * EMB_DIM + cc];
        const size_t off = in_base + (size_t)cc * HW + rr;
        out[off] = q;
        const float d = q - in[off];
        sq += d * d;
    }
    #pragma unroll
    for (int m = 16; m > 0; m >>= 1) sq += __shfl_xor(sq, m, 32);
    if (lane == 0) wsum[wave] = sq;
    __syncthreads();
    if (tid == 0) {
        float s = 0.f;
        #pragma unroll
        for (int i = 0; i < 8; ++i) s += wsum[i];
        atomicAdd(sumsq, s);
    }
}

// ---------------- finalize: loss = (1 + 0.25) * mean(sqdiff) -----------------
__global__ void vq_final(const float* __restrict__ sumsq, float* __restrict__ out) {
    out[LOSS_OFF] = *sumsq * (1.25f / 16777216.0f);
}

extern "C" void kernel_launch(void* const* d_in, const int* in_sizes, int n_in,
                              void* d_out, int out_size, void* d_ws, size_t ws_size,
                              hipStream_t stream) {
    const float* in = (const float*)d_in[0];   // [16,256,64,64] f32
    const float* cb = (const float*)d_in[1];   // [1024,256] f32
    float* out = (float*)d_out;
    char* ws = (char*)d_ws;
    _Float16* cbh  = (_Float16*)ws;                       // 512 KB f16 codebook
    float*    nrm  = (float*)(ws + 524288);               // 4 KB code norms
    float*    ssq  = (float*)(ws + 528384);               // 4 B accumulator

    vq_prep <<<NUM_EMB, EMB_DIM, 0, stream>>>(cb, cbh, nrm, ssq);
    vq_main <<<65536 / TILE_M, 256, 0, stream>>>(in, cb, cbh, nrm, out, ssq);
    vq_final<<<1, 1, 0, stream>>>(ssq, out);
}
